// HMM_9921374454107
// MI455X (gfx1250) — matrix operand order repaired and checked by reading, not compiled, not run on hardware
//
#include <hip/hip_runtime.h>

typedef __attribute__((ext_vector_type(16))) _Float16 v16h;
typedef __attribute__((ext_vector_type(8)))  _Float16 v8h;
typedef __attribute__((ext_vector_type(8)))  float    v8f;

#define NST 512
#define MSY 10000
#define BAT 64
#define TMX 256

// ---------------- kernel 1: log_softmax of priors (N=512), 1 WG ----------------
__global__ void k_priors(const float* __restrict__ u, float* __restrict__ lp) {
    __shared__ float red[512];
    int tid = threadIdx.x;
    float v = u[tid];
    red[tid] = v;
    __syncthreads();
    for (int s = 256; s > 0; s >>= 1) {
        if (tid < s) red[tid] = fmaxf(red[tid], red[tid + s]);
        __syncthreads();
    }
    float mx = red[0];
    __syncthreads();
    red[tid] = __expf(v - mx);
    __syncthreads();
    for (int s = 256; s > 0; s >>= 1) {
        if (tid < s) red[tid] += red[tid + s];
        __syncthreads();
    }
    float lz = mx + __logf(red[0]);
    lp[tid] = v - lz;
}

// ------------- kernel 2: per-row logZ of emissions (512 rows x 10000) -----------
__global__ void k_emlz(const float* __restrict__ em, float* __restrict__ lz) {
    __shared__ float red[256];
    int i = blockIdx.x, tid = threadIdx.x;
    const float* row = em + (size_t)i * MSY;
    float mx = -INFINITY;
    for (int c = tid; c < MSY; c += 256) mx = fmaxf(mx, row[c]);
    red[tid] = mx; __syncthreads();
    for (int s = 128; s > 0; s >>= 1) {
        if (tid < s) red[tid] = fmaxf(red[tid], red[tid + s]);
        __syncthreads();
    }
    mx = red[0]; __syncthreads();
    float sm = 0.f;
    for (int c = tid; c < MSY; c += 256) sm += __expf(row[c] - mx);
    red[tid] = sm; __syncthreads();
    for (int s = 128; s > 0; s >>= 1) {
        if (tid < s) red[tid] += red[tid + s];
        __syncthreads();
    }
    if (tid == 0) lz[i] = mx + __logf(red[0]);
}

// --- kernel 3: column softmax of trans (axis=0) -> f16  Tt[i][k] (u's layout) ----
__global__ void k_trans(const float* __restrict__ u, _Float16* __restrict__ Tt) {
    __shared__ float part[16][17];
    __shared__ float cmax[16], csum[16];
    int tx = threadIdx.x & 15;
    int ty = threadIdx.x >> 4;
    int k  = blockIdx.x * 16 + tx;
    float mx = -INFINITY;
    for (int r = 0; r < 32; ++r) {
        int i = ty + r * 16;
        mx = fmaxf(mx, u[(size_t)i * NST + k]);
    }
    part[ty][tx] = mx; __syncthreads();
    if (ty == 0) {
        float m2 = part[0][tx];
        for (int j = 1; j < 16; ++j) m2 = fmaxf(m2, part[j][tx]);
        cmax[tx] = m2;
    }
    __syncthreads();
    mx = cmax[tx];
    float sm = 0.f;
    for (int r = 0; r < 32; ++r) {
        int i = ty + r * 16;
        sm += __expf(u[(size_t)i * NST + k] - mx);
    }
    part[ty][tx] = sm; __syncthreads();
    if (ty == 0) {
        float s2 = 0.f;
        for (int j = 0; j < 16; ++j) s2 += part[j][tx];
        csum[tx] = s2;
    }
    __syncthreads();
    float rs = 1.0f / csum[tx];
    for (int r = 0; r < 32; ++r) {
        int i = ty + r * 16;
        Tt[(size_t)i * NST + k] = (_Float16)(__expf(u[(size_t)i * NST + k] - mx) * rs);
    }
}

// ------------------------- kernel 4: persistent scan ---------------------------
// 4 WGs x 512 threads (16 waves); WG owns 16 batch rows; 255 sequential steps.
// Scaled linear domain: alpha[b,i] = log(ul[b,i]) + C[b].  Two barriers/step:
//   phase A (all waves): S = P x Tt via dual-tile WMMA chain (K=512), emission
//     gathers issued before the chain; ul' = S * em_prob written to LDS.
//   phase B (wave w owns row b=w): row cached into 16 registers, max+sum via
//     shuffles, output capture, C/uinv update (wave-uniform regs), and the
//     normalized f16 Pa row written straight from the register cache.
__global__ void __launch_bounds__(512, 1) k_scan(
    const int*      __restrict__ x,
    const int*      __restrict__ Tlen,
    const float*    __restrict__ em,
    const float*    __restrict__ lzem,
    const float*    __restrict__ lprior,
    const _Float16* __restrict__ Tt,
    float*          __restrict__ out)
{
    __shared__ float    ul[16][NST + 4];     // linear-domain alpha (row stride 516)
    __shared__ _Float16 Pa[16][NST + 8];     // f16 A operand (row stride 520, 16B aligned)
    __shared__ int      sym[16];

    const int tid  = threadIdx.x;
    const int wave = tid >> 5;
    const int lane = tid & 31;
    const int bg0  = blockIdx.x * 16;

    const int tl = Tlen[bg0 + wave];         // wave-uniform: this row's length
    float c = 0.f;                           // running log-correction C[b=wave]
    float uinvv = 1.f;                       // 1/U for next normalize

    // ---- t = 0: alpha0[b,i] = log_em[i, x[b,0]] + log_priors[i] ----
    if (tid < 16) sym[tid] = x[(bg0 + tid) * TMX + 0];
    __syncthreads();
    for (int e = tid; e < 16 * NST; e += 512) {
        int b = e >> 9, i = e & (NST - 1);
        ul[b][i] = em[(size_t)i * MSY + sym[b]] - lzem[i] + lprior[i];
    }
    __syncthreads();
    {   // wave w: reduce row w, emit if T==1, write initial Pa row (max -> 1)
        float vals[16];
        float mx = -INFINITY;
        #pragma unroll
        for (int j = 0; j < 16; ++j) {
            vals[j] = ul[wave][lane + 32 * j];
            mx = fmaxf(mx, vals[j]);
        }
        for (int m = 16; m >= 1; m >>= 1) mx = fmaxf(mx, __shfl_xor(mx, m, 32));
        float sm = 0.f;
        #pragma unroll
        for (int j = 0; j < 16; ++j) {
            float e0 = __expf(vals[j] - mx);
            Pa[wave][lane + 32 * j] = (_Float16)e0;
            sm += e0;
        }
        for (int m = 16; m >= 1; m >>= 1) sm += __shfl_xor(sm, m, 32);
        c = mx;
        uinvv = 1.0f;
        if (lane == 0 && tl == 1) out[bg0 + wave] = mx + __logf(sm);
    }
    if (tid < 16) sym[tid] = x[(bg0 + tid) * TMX + 1];
    __syncthreads();

    for (int t = 1; t < TMX; ++t) {
        // ---- phase A: S = P x Tt via WMMA (2 i-tiles/wave, shared A) ----
        const int mr   = lane & 15;                // A-matrix row (batch)
        const int kb0  = (lane < 16) ? 0 : 8;      // A fragment K sub-block
        const int i0a  = wave * 16;                // first i-tile
        const int i0b  = i0a + 256;                // second i-tile
        const int ica  = i0a + (lane & 15);
        const int icb  = i0b + (lane & 15);
        const int brow = (lane >> 4) * 8;

        // prefetch emission values (scattered, L2-resident) before the WMMA chain
        float ea[8], eb[8];
        #pragma unroll
        for (int r = 0; r < 8; ++r) {
            int s = sym[brow + r];
            ea[r] = em[(size_t)ica * MSY + s];
            eb[r] = em[(size_t)icb * MSY + s];
        }
        const float lza = lzem[ica], lzb = lzem[icb];

        v8f acc0 = {}, acc1 = {};
        #pragma unroll 4
        for (int kc = 0; kc < 16; ++kc) {
            const int k0 = kc * 32;
            // A: 16x32 f16 fragment per ISA layout (two 16B LDS loads)
            v8h lo = *(const v8h*)&Pa[mr][k0 + kb0];
            v8h hi = *(const v8h*)&Pa[mr][k0 + kb0 + 16];
            v16h a = __builtin_shufflevector(lo, hi,
                0, 1, 2, 3, 4, 5, 6, 7, 8, 9, 10, 11, 12, 13, 14, 15);
            // B: 32x16 f16, lane = column i, K in same table order as A (two 16B loads)
            v8h b0l = *(const v8h*)&Tt[(size_t)ica * NST + k0 + kb0];
            v8h b0h = *(const v8h*)&Tt[(size_t)ica * NST + k0 + kb0 + 16];
            v8h b1l = *(const v8h*)&Tt[(size_t)icb * NST + k0 + kb0];
            v8h b1h = *(const v8h*)&Tt[(size_t)icb * NST + k0 + kb0 + 16];
            v16h b0 = __builtin_shufflevector(b0l, b0h,
                0, 1, 2, 3, 4, 5, 6, 7, 8, 9, 10, 11, 12, 13, 14, 15);
            v16h b1 = __builtin_shufflevector(b1l, b1h,
                0, 1, 2, 3, 4, 5, 6, 7, 8, 9, 10, 11, 12, 13, 14, 15);
            acc0 = __builtin_amdgcn_wmma_f32_16x16x32_f16(
                false, a, false, b0, (short)0, acc0, false, false);
            acc1 = __builtin_amdgcn_wmma_f32_16x16x32_f16(
                false, a, false, b1, (short)0, acc1, false, false);
        }

        // ul' = S * em_prob (stays linear; C untouched until phase B)
        #pragma unroll
        for (int r = 0; r < 8; ++r) {
            int b = brow + r;
            ul[b][ica] = acc0[r] * __expf(ea[r] - lza);
            ul[b][icb] = acc1[r] * __expf(eb[r] - lzb);
        }
        __syncthreads();

        // ---- phase B: wave w reduces + renormalizes its own row (fused) ----
        {
            float vals[16];
            float mx = 0.f, sm = 0.f;
            #pragma unroll
            for (int j = 0; j < 16; ++j) {
                vals[j] = ul[wave][lane + 32 * j];
                mx = fmaxf(mx, vals[j]);
                sm += vals[j];
            }
            for (int m = 16; m >= 1; m >>= 1) {
                mx = fmaxf(mx, __shfl_xor(mx, m, 32));
                sm += __shfl_xor(sm, m, 32);
            }
            if (lane == 0 && tl == t + 1) out[bg0 + wave] = __logf(sm) + c;
            float u = fmaxf(mx, 1e-30f);
            c += __logf(u);            // wave-uniform
            uinvv = 1.0f / u;
            #pragma unroll
            for (int j = 0; j < 16; ++j) {
                Pa[wave][lane + 32 * j] = (_Float16)(vals[j] * uinvv);
            }
        }
        if (tid < 16 && t + 1 < TMX) sym[tid] = x[(bg0 + tid) * TMX + (t + 1)];
        __syncthreads();
    }
}

extern "C" void kernel_launch(void* const* d_in, const int* in_sizes, int n_in,
                              void* d_out, int out_size, void* d_ws, size_t ws_size,
                              hipStream_t stream) {
    const int*   x       = (const int*)d_in[0];       // (B, T_MAX)
    const int*   Tlen    = (const int*)d_in[1];       // (B,)
    const float* upri    = (const float*)d_in[2];     // (N,)
    const float* utrans  = (const float*)d_in[3];     // (N, N)
    const float* uemit   = (const float*)d_in[4];     // (N, M)
    float*       out     = (float*)d_out;             // (B, 1)

    float*    lprior = (float*)d_ws;                          // 512 f32
    float*    lzem   = lprior + NST;                          // 512 f32
    _Float16* Tt     = (_Float16*)((char*)d_ws + 4096);       // 512*512 f16

    k_priors<<<1, 512, 0, stream>>>(upri, lprior);
    k_emlz  <<<NST, 256, 0, stream>>>(uemit, lzem);
    k_trans <<<NST / 16, 256, 0, stream>>>(utrans, Tt);
    k_scan  <<<BAT / 16, 512, 0, stream>>>(x, Tlen, uemit, lzem, lprior, Tt, out);
}
